// flownet_1752346656961
// MI455X (gfx1250) — compile-verified, matrix operand order checked by reading, not run on hardware
//
#include <hip/hip_runtime.h>
#include <math.h>

#define PI_F 3.14159265358979323846f

typedef __attribute__((ext_vector_type(16))) _Float16 v16h;
typedef __attribute__((ext_vector_type(8)))  float    v8f;

static __device__ __forceinline__ int iclamp(int x, int lo, int hi) {
  return x < lo ? lo : (x > hi ? hi : x);
}

// ---------------------------------------------------------------------------
// Per-layer gnomonic sampling grid: rows[Ho*K], dcol[Ho*K]
// ---------------------------------------------------------------------------
__global__ void grid_kernel(float* __restrict__ rows, float* __restrict__ dcol,
                            int Ho, int H, int W, int k, int stride, float delta) {
  int KK = k * k;
  int idx = blockIdx.x * blockDim.x + threadIdx.x;
  if (idx >= Ho * KK) return;
  int ho = idx / KK, kk = idx % KK;
  int iy = kk / k, ix = kk % k;
  float halfk = (k - 1) * 0.5f;
  float ty = tanf(delta / k * ((float)iy - halfk));
  float tx = tanf(delta / k * ((float)ix - halfk));
  float yy = ty, xx = tx;
  float rho = sqrtf(xx * xx + yy * yy);
  float rho_s = (rho == 0.f) ? 1.f : rho;
  float nu = atanf(rho);
  float lat0 = PI_F * 0.5f - ((float)(ho * stride) + 0.5f) * (PI_F / (float)H);
  float sl = sinf(lat0), cl = cosf(lat0);
  float sn = sinf(nu),  cn = cosf(nu);
  float v = cn * sl + (yy / rho_s) * sn * cl;
  v = fminf(1.f, fmaxf(-1.f, v));
  float lat = asinf(v);
  float dl = atan2f(xx * sn, rho * cl * cn - yy * sl * sn);
  rows[idx] = (PI_F * 0.5f - lat) / PI_F * (float)H - 0.5f;
  dcol[idx] = dl / (2.f * PI_F) * (float)W;
}

// ---------------------------------------------------------------------------
// Weights fp32 -> f16, K-dim padded to multiple of 64
// ---------------------------------------------------------------------------
__global__ void wcvt_kernel(const float* __restrict__ w, _Float16* __restrict__ wf,
                            int O, int Kd, int KdP) {
  int idx = blockIdx.x * blockDim.x + threadIdx.x;
  if (idx >= O * KdP) return;
  int o = idx / KdP, kd = idx % KdP;
  wf[idx] = (kd < Kd) ? (_Float16)w[(size_t)o * Kd + kd] : (_Float16)0.f;
}

// kd -> (channel, tap) lookup table: avoids per-element integer division by KK
__global__ void ckk_kernel(int* __restrict__ ckk, int Kd, int KdP, int KK) {
  int idx = blockIdx.x * blockDim.x + threadIdx.x;
  if (idx >= KdP) return;
  int c = idx / KK, kk = idx % KK;
  ckk[idx] = (idx < Kd) ? ((c << 16) | kk) : 0;
}

// ---------------------------------------------------------------------------
// Fused bilinear-gather + WMMA GEMM.
//  grid.x = Nsp/32 spatial tiles (N=32); block = (O/16) waves * 32 threads.
//  Phase 0 (once per block): bilinear table per (ncol, kk) -> LDS
//  Phase 1 (per 64-wide K chunk): gather 64x32 f16 tile via table, then each
//    wave issues 4 x v_wmma_f32_16x16x32_f16 into two fp32 accumulators.
//  All plane dims are powers of two: addresses are shift/OR only.
// ---------------------------------------------------------------------------
__global__ void __launch_bounds__(512)
sconv_wmma_kernel(const float* __restrict__ s1, const float* __restrict__ s2,
                  int C1, int C2, int shiftc,
                  int H, int Ho, int Wo, int KK, int Kd, int KdP,
                  int O, int stride,
                  int hwSh, int woSh, int Wm,          // log2(Ho*Wo), log2(Wo), W-1
                  int planeSh1, int wSh1,              // log2(Hs*Ws), log2(Ws) for s1
                  int planeSh2, int wSh2,              // log2(H*W),  log2(W)  for s2
                  const float* __restrict__ rows, const float* __restrict__ dcol,
                  const int* __restrict__ ckk,
                  const _Float16* __restrict__ wf, const float* __restrict__ bias,
                  float* __restrict__ out) {
  extern __shared__ char smem[];
  int4*     tabI = (int4*)smem;                                   // 32*KK entries
  float2*   tabW = (float2*)(smem + (size_t)32 * KK * sizeof(int4));
  _Float16* tb   = (_Float16*)(smem + (size_t)32 * KK * 24);      // 64 x 32 halves

  const int tid   = threadIdx.x;
  const int wave  = tid >> 5;
  const int lane  = tid & 31;
  const int nbase = blockIdx.x * 32;
  const int HoWoM = (1 << hwSh) - 1;
  const int WoM   = (1 << woSh) - 1;

  // Phase 0: per-(ncol, kk) bilinear indices/weights (channel-independent)
  for (int j = tid; j < 32 * KK; j += blockDim.x) {
    int ncol = j & 31, kk = j >> 5;
    int n   = nbase + ncol;
    int rem = n & HoWoM;
    int ho  = rem >> woSh, wo = rem & WoM;
    float rr = rows[ho * KK + kk];
    float cc = (float)(wo * stride) + dcol[ho * KK + kk];
    float r0f = floorf(rr), c0f = floorf(cc);
    int r0i = iclamp((int)r0f, 0, H - 1);
    int r1i = iclamp((int)r0f + 1, 0, H - 1);
    int c0i = (int)c0f & Wm;          // W power of two: wrap == AND
    int c1i = (c0i + 1) & Wm;
    tabI[j] = make_int4(r0i, r1i, c0i, c1i);
    tabW[j] = make_float2(rr - r0f, cc - c0f);
  }
  __syncthreads();

  v8f acc0 = {};
  v8f acc1 = {};

  for (int kc = 0; kc < KdP; kc += 64) {
    // gather 64x32 f16 patch tile into LDS using the cached table
    for (int i = tid; i < 64 * 32; i += blockDim.x) {
      int kd = kc + (i >> 5);
      float val = 0.f;
      if (kd < Kd) {
        int pk = ckk[kd];
        int c  = pk >> 16, kk = pk & 0xffff;
        int e  = (kk << 5) | (i & 31);
        int4   rc = tabI[e];
        float2 wv = tabW[e];
        int b = (nbase + (i & 31)) >> hwSh;
        float v00, v01, v10, v11;
        if (c < C1) {   // up2() source: coords shift down, smaller plane
          const float* p = s1 + (((size_t)b * C1 + c) << planeSh1);
          int r0 = rc.x >> shiftc, r1 = rc.y >> shiftc;
          int c0 = rc.z >> shiftc, c1 = rc.w >> shiftc;
          v00 = p[(r0 << wSh1) | c0];
          v01 = p[(r0 << wSh1) | c1];
          v10 = p[(r1 << wSh1) | c0];
          v11 = p[(r1 << wSh1) | c1];
        } else {
          const float* p = s2 + (((size_t)b * C2 + (c - C1)) << planeSh2);
          v00 = p[(rc.x << wSh2) | rc.z];
          v01 = p[(rc.x << wSh2) | rc.w];
          v10 = p[(rc.y << wSh2) | rc.z];
          v11 = p[(rc.y << wSh2) | rc.w];
        }
        float wr = wv.x, wc = wv.y;
        val = v00 * (1.f - wr) * (1.f - wc) + v01 * (1.f - wr) * wc
            + v10 * wr * (1.f - wc)         + v11 * wr * wc;
      }
      tb[i] = (_Float16)val;
    }
    __syncthreads();

    // 4 WMMAs per LDS tile: 2 K-subchunks x 2 N-subtiles
    {
      int m  = lane & 15;
      int hi = lane >> 4;   // lane<16: K {0..7,16..23}; lane>=16: {8..15,24..31}
      int nn = lane & 15;
#pragma unroll
      for (int s = 0; s < 2; ++s) {
        const _Float16* wrow =
            &wf[(size_t)(wave * 16 + m) * KdP + kc + s * 32 + hi * 8];
        v16h a;
#pragma unroll
        for (int i = 0; i < 8; ++i) { a[i] = wrow[i]; a[i + 8] = wrow[16 + i]; }

        v16h b0, b1;
#pragma unroll
        for (int i = 0; i < 8; ++i) {
          int kr0 = s * 32 + hi * 8 + i;
          int kr1 = s * 32 + hi * 8 + 16 + i;
          b0[i]     = tb[kr0 * 32 + nn];
          b0[i + 8] = tb[kr1 * 32 + nn];
          b1[i]     = tb[kr0 * 32 + 16 + nn];
          b1[i + 8] = tb[kr1 * 32 + 16 + nn];
        }
        acc0 = __builtin_amdgcn_wmma_f32_16x16x32_f16(
            false, a, false, b0, (short)0, acc0, false, false);
        acc1 = __builtin_amdgcn_wmma_f32_16x16x32_f16(
            false, a, false, b1, (short)0, acc1, false, false);
      }
    }
    __syncthreads();
  }

  // C/D layout: VGPR r -> M = r + 8*(lane>=16), N = lane&15
  {
    int nl = lane & 15, mo = (lane >> 4) * 8;
#pragma unroll
    for (int t = 0; t < 2; ++t) {
      int n   = nbase + t * 16 + nl;
      int b   = n >> hwSh;
      int rem = n & HoWoM;
      int ho  = rem >> woSh, wo = rem & WoM;
#pragma unroll
      for (int r = 0; r < 8; ++r) {
        int o = wave * 16 + mo + r;
        float v = (t == 0) ? acc0[r] : acc1[r];
        out[(((size_t)b * O + o) * Ho + ho) * Wo + wo] = v + bias[o];
      }
    }
  }
}

// ---------------------------------------------------------------------------
// Train-mode BN: per-channel stats folded into scale/shift, then ReLU in place
// ---------------------------------------------------------------------------
__global__ void bn_stats_kernel(const float* __restrict__ x, const float* __restrict__ g,
                                const float* __restrict__ bb,
                                float* __restrict__ scale, float* __restrict__ shift,
                                int O, int HoWo, int B) {
  __shared__ float ssum[256], ssq[256];
  int o = blockIdx.x, tid = threadIdx.x;
  int Nsp = B * HoWo;
  float s = 0.f, q = 0.f;
  for (int j = tid; j < Nsp; j += 256) {
    int b = j / HoWo, sp = j % HoWo;
    float v = x[((size_t)b * O + o) * HoWo + sp];
    s += v; q += v * v;
  }
  ssum[tid] = s; ssq[tid] = q;
  __syncthreads();
  for (int st = 128; st > 0; st >>= 1) {
    if (tid < st) { ssum[tid] += ssum[tid + st]; ssq[tid] += ssq[tid + st]; }
    __syncthreads();
  }
  if (tid == 0) {
    float m   = ssum[0] / (float)Nsp;
    float var = ssq[0] / (float)Nsp - m * m;
    float inv = rsqrtf(var + 1e-5f);
    scale[o] = g[o] * inv;
    shift[o] = bb[o] - m * g[o] * inv;
  }
}

__global__ void bn_apply_kernel(float* __restrict__ x, const float* __restrict__ scale,
                                const float* __restrict__ shift,
                                int O, int HoWo, long total) {
  long idx = (long)blockIdx.x * blockDim.x + threadIdx.x;
  if (idx >= total) return;
  int o = (int)((idx / HoWo) % O);
  float y = x[idx] * scale[o] + shift[o];
  x[idx] = fmaxf(y, 0.f);
}

// ---------------------------------------------------------------------------
// Host launch
// ---------------------------------------------------------------------------
extern "C" void kernel_launch(void* const* d_in, const int* in_sizes, int n_in,
                              void* d_out, int out_size, void* d_ws, size_t ws_size,
                              hipStream_t stream) {
  (void)in_sizes; (void)n_in; (void)out_size; (void)ws_size;
  const int B = 4;
  char* ws = (char*)d_ws;
  size_t off = 0;
  auto alloc = [&](size_t bytes) -> void* {
    void* p = ws + off;
    off += (bytes + 255) & ~(size_t)255;
    return p;
  };
  float*    rows  = (float*)alloc(8192 * sizeof(float));
  float*    dcol  = (float*)alloc(8192 * sizeof(float));
  _Float16* wf    = (_Float16*)alloc((size_t)256 * 3456 * sizeof(_Float16));
  int*      ckk   = (int*)alloc(4096 * sizeof(int));
  float*    scale = (float*)alloc(256 * sizeof(float));
  float*    shift = (float*)alloc(256 * sizeof(float));
  float* x1  = (float*)alloc((size_t)4 * 32  * 128 * 256 * 4);
  float* x2  = (float*)alloc((size_t)4 * 64  * 64  * 128 * 4);
  float* x3  = (float*)alloc((size_t)4 * 128 * 32  * 64  * 4);
  float* x31 = (float*)alloc((size_t)4 * 128 * 32  * 64  * 4);
  float* x4  = (float*)alloc((size_t)4 * 256 * 16  * 32  * 4);
  float* x41 = (float*)alloc((size_t)4 * 256 * 16  * 32  * 4);
  float* x5  = (float*)alloc((size_t)4 * 256 * 32  * 64  * 4);

  const float* input = (const float*)d_in[0];

  struct L {
    const float *s1, *s2;
    int C1, C2, shiftc, H, W, O, k, stride;
    float delta;
    int widx;
    float* dst;
  };
  L Ls[8] = {
    { input, input, 6,   0,   0, 256, 512,  32, 7, 2, PI_F / 32.f, 1,  x1  },
    { x1,    x1,    32,  0,   0, 128, 256,  64, 5, 2, PI_F / 16.f, 5,  x2  },
    { x2,    x2,    64,  0,   0, 64,  128, 128, 5, 2, PI_F / 4.f,  9,  x3  },
    { x3,    x3,    128, 0,   0, 32,  64,  128, 3, 1, PI_F / 4.f,  13, x31 },
    { x31,   x31,   128, 0,   0, 32,  64,  256, 3, 2, PI_F / 2.f,  17, x4  },
    { x4,    x4,    256, 0,   0, 16,  32,  256, 3, 1, PI_F / 2.f,  21, x41 },
    { x41,   x31,   256, 128, 1, 32,  64,  256, 3, 1, PI_F / 4.f,  25, x5  },
    { x5,    x2,    256, 64,  1, 64,  128, 128, 3, 1, PI_F / 8.f,  29, (float*)d_out },
  };

  for (int li = 0; li < 8; ++li) {
    const L& P = Ls[li];
    int KK  = P.k * P.k;
    int C   = P.C1 + P.C2;
    int Kd  = C * KK;
    int KdP = (Kd + 63) & ~63;
    int Ho  = P.H / P.stride, Wo = P.W / P.stride;
    int Nsp = B * Ho * Wo;
    int hwSh = __builtin_ctz(Ho * Wo);
    int woSh = __builtin_ctz(Wo);
    int wSh2 = __builtin_ctz(P.W);
    int planeSh2 = __builtin_ctz(P.H * P.W);
    int wSh1 = wSh2 - P.shiftc;
    int planeSh1 = planeSh2 - 2 * P.shiftc;
    const float* w    = (const float*)d_in[P.widx + 0];
    const float* bias = (const float*)d_in[P.widx + 1];
    const float* g    = (const float*)d_in[P.widx + 2];
    const float* bb   = (const float*)d_in[P.widx + 3];

    int gt = Ho * KK;
    grid_kernel<<<(gt + 255) / 256, 256, 0, stream>>>(rows, dcol, Ho, P.H, P.W,
                                                      P.k, P.stride, P.delta);
    int wt = P.O * KdP;
    wcvt_kernel<<<(wt + 255) / 256, 256, 0, stream>>>(w, wf, P.O, Kd, KdP);
    ckk_kernel<<<(KdP + 255) / 256, 256, 0, stream>>>(ckk, Kd, KdP, KK);

    dim3 gg(Nsp / 32);
    dim3 blk((P.O / 16) * 32);
    size_t shmem = (size_t)32 * KK * 24 + (size_t)64 * 32 * 2;
    sconv_wmma_kernel<<<gg, blk, shmem, stream>>>(P.s1, P.s2, P.C1, P.C2, P.shiftc,
        P.H, Ho, Wo, KK, Kd, KdP, P.O, P.stride,
        hwSh, woSh, P.W - 1, planeSh1, wSh1, planeSh2, wSh2,
        rows, dcol, ckk, wf, bias, P.dst);

    bn_stats_kernel<<<P.O, 256, 0, stream>>>(P.dst, g, bb, scale, shift, P.O, Ho * Wo, B);
    long total = (long)B * P.O * Ho * Wo;
    bn_apply_kernel<<<(unsigned)((total + 255) / 256), 256, 0, stream>>>(
        P.dst, scale, shift, P.O, Ho * Wo, total);
  }
}
